// GCNLayer_15358803050718
// MI455X (gfx1250) — matrix operand order repaired and checked by reading, not compiled, not run on hardware
//
#include <hip/hip_runtime.h>
#include <hip/hip_bf16.h>
#include <math.h>

typedef __attribute__((ext_vector_type(16))) __bf16 v16bf;
typedef __attribute__((ext_vector_type(8)))  __bf16 v8bf;
typedef __attribute__((ext_vector_type(4)))  __bf16 v4bf;
typedef __attribute__((ext_vector_type(8)))  float  v8f;
typedef __attribute__((ext_vector_type(4)))  float  f32x4;

#define IN_FEATS  128
#define OUT_FEATS 128
#define LDS_PAD   8                         // bf16 elements of row padding
#define LDS_STRIDE (IN_FEATS + LDS_PAD)     // 136 bf16 = 272 B per row

__device__ __forceinline__ __bf16 f2bf(float f) { return (__bf16)f; }

// Fast tanh on the transcendental unit: tanh(x) = sign(x)*(1 - 2/(e^{2|x|}+1)).
__device__ __forceinline__ float fast_tanh(float x) {
  float ax = fabsf(x);
  float e  = __expf(2.0f * ax);            // v_exp_f32
  float t  = 1.0f - 2.0f / (e + 1.0f);     // v_rcp_f32 + fma
  return copysignf(t, x);
}

// Pack two contiguous 8-float chunks into a v16bf (elements 0..7 from p0, 8..15 from p1).
__device__ __forceinline__ v16bf pack8x2_bf(const float* __restrict__ p0,
                                            const float* __restrict__ p1) {
  const f32x4* q0 = (const f32x4*)p0;
  const f32x4* q1 = (const f32x4*)p1;
  f32x4 c0 = q0[0], c1 = q0[1], c2 = q1[0], c3 = q1[1];
  v16bf r;
#pragma unroll
  for (int i = 0; i < 4; ++i) {
    r[i]      = f2bf(c0[i]);
    r[4 + i]  = f2bf(c1[i]);
    r[8 + i]  = f2bf(c2[i]);
    r[12 + i] = f2bf(c3[i]);
  }
  return r;
}

// ---------------------------------------------------------------------------
// Kernel 0: zero the aggregation workspace (harness poisons d_ws with 0xAA).
// ---------------------------------------------------------------------------
__global__ void gcn_zero_kernel(float* __restrict__ agg, int n4) {
  int i = blockIdx.x * blockDim.x + threadIdx.x;
  f32x4 z = {0.f, 0.f, 0.f, 0.f};
  if (i < n4) ((f32x4*)agg)[i] = z;
}

// ---------------------------------------------------------------------------
// Kernel 1: edge scatter-add (SpMM).  One wave32 per edge: each lane moves a
// float4 of the source-node feature and atomically adds it to the dst row.
// feature (25.6 MB) and agg (25.6 MB) are both L2-resident on MI455X (192 MB).
// ---------------------------------------------------------------------------
__global__ void gcn_scatter_kernel(const float* __restrict__ feature,
                                   const int*   __restrict__ src,
                                   const int*   __restrict__ dst,
                                   float*       __restrict__ agg,
                                   int n_edges) {
  const int lane = threadIdx.x & 31;
  const int wave = threadIdx.x >> 5;
  const int e = blockIdx.x * (blockDim.x >> 5) + wave;
  if (e >= n_edges) return;

  const int s = src[e];
  const int d = dst[e];

  const f32x4 v = ((const f32x4*)(feature + (size_t)s * IN_FEATS))[lane];
  float* o = agg + (size_t)d * IN_FEATS + lane * 4;
  unsafeAtomicAdd(o + 0, v[0]);   // global_atomic_add_f32
  unsafeAtomicAdd(o + 1, v[1]);
  unsafeAtomicAdd(o + 2, v[2]);
  unsafeAtomicAdd(o + 3, v[3]);
}

// ---------------------------------------------------------------------------
// Kernel 2: out = tanh(agg @ W^T + b) via v_wmma_f32_16x16x32_bf16.
//  * Block = 256 threads (8 waves).  The block cooperatively converts the
//    whole W (128x128 f32 -> bf16) into padded LDS ONCE, then every wave
//    feeds its B tiles from LDS with ds_load_b128 (no redundant cvt/global).
//  * Each wave owns one 16-row M-tile and all 8 N-tiles (8 v8f accumulators),
//    K = 128 in 4 steps of 32.
// ---------------------------------------------------------------------------
__global__ void gcn_gemm_tanh_kernel(const float* __restrict__ agg,
                                     const float* __restrict__ W,     // [128][128]
                                     const float* __restrict__ bias,  // [128]
                                     float*       __restrict__ out,
                                     int n_rows) {
  __shared__ __bf16 Wlds[OUT_FEATS * LDS_STRIDE];   // 34 KB of the 320 KB WGP LDS

  // --- Cooperative W f32->bf16 staging (all 256 threads; before any return) ---
  // 128 rows x 32 quads of 4 floats; each thread converts 16 quads.
  for (int q = threadIdx.x; q < OUT_FEATS * (IN_FEATS / 4); q += blockDim.x) {
    const int row = q >> 5;            // IN_FEATS/4 == 32 quads per row
    const int c4  = (q & 31) * 4;
    const f32x4 v = ((const f32x4*)(W + (size_t)row * IN_FEATS))[q & 31];
    v4bf p;
#pragma unroll
    for (int i = 0; i < 4; ++i) p[i] = f2bf(v[i]);
    *(v4bf*)&Wlds[row * LDS_STRIDE + c4] = p;        // ds_store_b64
  }
  __syncthreads();

  const int lane = threadIdx.x & 31;
  const int wave = threadIdx.x >> 5;
  const int mtile = blockIdx.x * 8 + wave;          // wave-uniform guard after
  if (mtile * 16 >= n_rows) return;                 // barrier: EXEC all-1s for WMMA

  const int colid = lane & 15;
  const int half  = lane >> 4;                      // 0 or 1

  int m = mtile * 16 + colid;                       // A row owned by this lane
  if (m >= n_rows) m = n_rows - 1;                  // clamp loads (stores guarded)
  const float* arow = agg + (size_t)m * IN_FEATS;

  v8f acc[8];
#pragma unroll
  for (int nt = 0; nt < 8; ++nt)
#pragma unroll
    for (int r = 0; r < 8; ++r) acc[nt][r] = 0.f;

#pragma unroll
  for (int kk = 0; kk < IN_FEATS; kk += 32) {
    // A tile (16x32 bf16): lane holds 16 contiguous K at offset half*16.
    const float* pa = arow + kk + half * 16;
    const v16bf a = pack8x2_bf(pa, pa + 8);

#pragma unroll
    for (int nt = 0; nt < 8; ++nt) {
      // B = W^T: lane holds W row (n), 16 contiguous bf16 K at offset half*16,
      // fetched from LDS (16-B aligned -> 2x ds_load_b128).
      const int n = nt * 16 + colid;
      const v8bf* lp = (const v8bf*)&Wlds[n * LDS_STRIDE + kk + half * 16];
      const v8bf b0 = lp[0];
      const v8bf b1 = lp[1];
      v16bf b;
#pragma unroll
      for (int i = 0; i < 8; ++i) { b[i] = b0[i]; b[8 + i] = b1[i]; }
      acc[nt] = __builtin_amdgcn_wmma_f32_16x16x32_bf16(
          /*neg_a=*/false, a, /*neg_b=*/false, b,
          /*c_mod=*/(short)0, acc[nt], /*reuse_a=*/false, /*reuse_b=*/false);
    }
  }

  // Epilogue: D VGPR r -> row (mtile*16 + half*8 + r), col (nt*16 + colid).
  const int rbase = mtile * 16 + half * 8;
#pragma unroll
  for (int nt = 0; nt < 8; ++nt) {
    const float bv = bias[nt * 16 + colid];
#pragma unroll
    for (int r = 0; r < 8; ++r) {
      const int row = rbase + r;
      if (row < n_rows)
        out[(size_t)row * OUT_FEATS + nt * 16 + colid] = fast_tanh(acc[nt][r] + bv);
    }
  }
}

// ---------------------------------------------------------------------------
extern "C" void kernel_launch(void* const* d_in, const int* in_sizes, int n_in,
                              void* d_out, int out_size, void* d_ws, size_t ws_size,
                              hipStream_t stream) {
  const float* feature = (const float*)d_in[0];   // [N, 128]
  const float* W       = (const float*)d_in[1];   // [128, 128]
  const float* bias    = (const float*)d_in[2];   // [128]
  const int*   src     = (const int*)d_in[3];     // [E]
  const int*   dst     = (const int*)d_in[4];     // [E]
  float*       out     = (float*)d_out;           // [N, 128]

  const int n_rows  = in_sizes[0] / IN_FEATS;     // 50000
  const int n_edges = in_sizes[3];                // 600000

  float* agg = (float*)d_ws;                      // [N, 128] f32 scratch

  // 1) zero agg
  {
    const int n4 = n_rows * IN_FEATS / 4;
    const int threads = 256;
    const int blocks = (n4 + threads - 1) / threads;
    gcn_zero_kernel<<<blocks, threads, 0, stream>>>(agg, n4);
  }
  // 2) scatter-add over edges (8 waves / 256-thread block, one edge per wave)
  {
    const int threads = 256;
    const int waves_per_block = threads / 32;
    const int blocks = (n_edges + waves_per_block - 1) / waves_per_block;
    gcn_scatter_kernel<<<blocks, threads, 0, stream>>>(feature, src, dst, agg, n_edges);
  }
  // 3) WMMA GEMM + bias + tanh (8 M-tiles of 16 rows per 256-thread block)
  {
    const int threads = 256;
    const int mtiles = (n_rows + 15) / 16;
    const int blocks = (mtiles + 7) / 8;
    gcn_gemm_tanh_kernel<<<blocks, threads, 0, stream>>>(agg, W, bias, out, n_rows);
  }
}
